// RHN_audio_encoder_10746008175090
// MI455X (gfx1250) — matrix operand order repaired and checked by reading, not compiled, not run on hardware
//
#include <hip/hip_runtime.h>
#include <hip/hip_bf16.h>
#include <cstdint>

// ---------------------------------------------------------------------------
// RHN audio encoder for gfx1250 (MI455X).
//  * all GEMMs via v_wmma_f32_16x16x32_bf16 (bf16 in, fp32 accum)
//  * K-loops software-pipelined so waitcnt can be partial (overlap
//    global_load_b128 stream with the WMMA stream); 2-deep in the scan
//  * recurrent scan: persistent kernel (32 blocks x 4 waves), one 16x16 tile
//    per wave, fp32 state in registers, state broadcast staged into LDS by
//    the Tensor Data Mover (tensor_load_to_lds + s_wait_tensorcnt), with an
//    asm escape/clobber so the compiler keeps the LDS reads of DMA-written
//    data; device-wide software barrier between sub-steps
// ---------------------------------------------------------------------------

typedef __bf16 bf16_t;
typedef __attribute__((ext_vector_type(16))) __bf16 v16bf;
typedef __attribute__((ext_vector_type(8)))  float  v8f;
typedef __attribute__((ext_vector_type(4)))  unsigned int v4u;
typedef __attribute__((ext_vector_type(8)))  int v8i;
typedef __attribute__((ext_vector_type(4)))  int v4i;

#define HID   1024
#define BATCH 32
#define TLEN  1024
#define DEPTH 2
#define SCAN_BLOCKS 32

#if defined(__HIP_DEVICE_COMPILE__) && __has_builtin(__builtin_amdgcn_tensor_load_to_lds) && __has_builtin(__builtin_amdgcn_s_wait_tensorcnt)
#define USE_TDM 1
#else
#define USE_TDM 0
#endif

struct bytes32 { uint4 a, b; };

__device__ __forceinline__ v8f zero8() {
  v8f v; for (int i = 0; i < 8; ++i) v[i] = 0.f; return v;
}

__device__ __forceinline__ v8f wmma_bf16(v16bf a, v16bf b, v8f c) {
  return __builtin_amdgcn_wmma_f32_16x16x32_bf16(false, a, false, b,
                                                 (short)0, c, false, false);
}

// The TDM writes LDS behind the compiler's back: escape the pointer through
// an asm with a memory clobber so loads of the DMA'd data are not folded to
// undef / const-propagated across the transfer.
__device__ __forceinline__ bf16_t* lds_acquire(bf16_t* p) {
#if defined(__HIP_DEVICE_COMPILE__)
  asm volatile("" : "+v"(p) :: "memory");
#endif
  return p;
}

// A fragment (16x32 bf16) per ISA layout:
//   lanes 0-15 : M=lane,    K = {k0..k0+7} and {k0+16..k0+23}, k0 = kb*32
//   lanes16-31 : M=lane-16, K = {k0+8..k0+15} and {k0+24..k0+31}
__device__ __forceinline__ v16bf frag2(const bf16_t* p0, const bf16_t* p1) {
  bytes32 t;
  t.a = *(const uint4*)p0;
  t.b = *(const uint4*)p1;
  return __builtin_bit_cast(v16bf, t);
}

__device__ __forceinline__ v16bf load_A_lds(const bf16_t* sA, int m_tile, int kb) {
  int lane = threadIdx.x & 31;
  const bf16_t* base = sA + (size_t)(m_tile * 16 + (lane & 15)) * HID
                          + kb * 32 + ((lane & 16) ? 8 : 0);
  return frag2(base, base + 16);
}

__device__ __forceinline__ v16bf load_A_g(const bf16_t* X, int ldk, int m_tile, int kb) {
  int lane = threadIdx.x & 31;
  const bf16_t* base = X + (size_t)(m_tile * 16 + (lane & 15)) * ldk
                         + kb * 32 + ((lane & 16) ? 8 : 0);
  return frag2(base, base + 16);
}

// B packed layout: tile (n_tile, kb): 32 lanes x 16 bf16 contiguous (32B/lane)
__device__ __forceinline__ v16bf load_B_p(const bf16_t* Bp, int n_tile, int kb, int nkb) {
  int lane = threadIdx.x & 31;
  const bf16_t* base = Bp + (((size_t)n_tile * nkb + kb) * 32 + lane) * 16;
  bytes32 t = *(const bytes32*)base;
  return __builtin_bit_cast(v16bf, t);
}

__device__ __forceinline__ float fast_tanh(float x) {
  float e2 = __expf(2.f * x);
  return (e2 - 1.f) / (e2 + 1.f);
}
__device__ __forceinline__ float fast_sigmoid(float x) {
  return 1.f / (1.f + __expf(-x));
}

// device-wide barrier for the persistent scan kernel (SCAN_BLOCKS blocks)
__device__ __forceinline__ void grid_sync(unsigned* bar) {
  __syncthreads();
  if (threadIdx.x == 0) {
    __threadfence();
    unsigned gen = atomicAdd(bar + 1, 0u);
    if (atomicAdd(bar, 1u) == (unsigned)(SCAN_BLOCKS - 1)) {
      atomicExch(bar, 0u);
      __threadfence();
      atomicAdd(bar + 1, 1u);
    } else {
      while (atomicAdd(bar + 1, 0u) == gen) __builtin_amdgcn_s_sleep(1);
    }
    __threadfence();
  }
  __syncthreads();
}

#if USE_TDM
// TDM: stage the 64 KB bf16 state [32 x 1024] (viewed as 16 rows x 2048 elem,
// 2B each, contiguous) from global into LDS offset 0.  D# per ISA 08 §8.3/8.4.
__device__ __forceinline__ void tdm_stage_state(const bf16_t* gsrc) {
  unsigned long long ga = (unsigned long long)(uintptr_t)gsrc;
  v4u g0;
  g0[0] = 1u;                                   // count=1 (valid user D#)
  g0[1] = 0u;                                   // lds_addr = 0 (sA at base)
  g0[2] = (unsigned)ga;                         // global_addr[31:0]
  g0[3] = (unsigned)((ga >> 32) & 0x01ffffffu)  // global_addr[56:32]
        | (2u << 30);                           // type = 2 (image)
  v8i g1;
  g1[0] = (int)(1u << 16);        // workgroup_mask=0, data_size=1 (2 bytes)
  g1[1] = (int)(2048u << 16);     // tensor_dim0[15:0] = 2048
  g1[2] = (int)(16u << 16);       // tensor_dim0[31:16]=0 | tensor_dim1 = 16
  g1[3] = (int)(2048u << 16);     // tensor_dim1[31:16]=0 | tile_dim0 = 2048
  g1[4] = 16;                     // tile_dim1 = 16, tile_dim2 = 0
  g1[5] = 2048;                   // tensor_dim0_stride[31:0] = 2048
  g1[6] = 0;                      // stride hi | tensor_dim1_stride lo
  g1[7] = 0;
  v4i z4; z4[0] = z4[1] = z4[2] = z4[3] = 0;    // groups 2/3 unused (2D)
#if __clang_major__ >= 23
  v8i z8; for (int i = 0; i < 8; ++i) z8[i] = 0;
  __builtin_amdgcn_tensor_load_to_lds(g0, g1, z4, z4, z8, 0);
#else
  __builtin_amdgcn_tensor_load_to_lds(g0, g1, z4, z4, 0);
#endif
}
#endif

// ---------------------------------------------------------------------------
// fp32 [K,N] -> packed bf16 WMMA-B tiles
// ---------------------------------------------------------------------------
__global__ void pack_b_kernel(const float* __restrict__ src, bf16_t* __restrict__ dst,
                              int K, int N) {
  int idx = blockIdx.x * blockDim.x + threadIdx.x;
  int nkb = K / 32;
  int total = nkb * (N / 16) * 32;
  if (idx >= total) return;
  int lane   = idx & 31;
  int group  = idx >> 5;
  int kb     = group % nkb;
  int n_tile = group / nkb;
  int n  = n_tile * 16 + (lane & 15);
  int k0 = kb * 32 + ((lane & 16) ? 8 : 0);
  bf16_t* d = dst + (size_t)idx * 16;
  for (int j = 0; j < 16; ++j) d[j] = (bf16_t)src[(size_t)(k0 + j + ((j >= 8) ? 8 : 0)) * N + n];
}

// ---------------------------------------------------------------------------
// conv (40x6, stride 2 on W, valid) -> X0 bf16 [T=1024][B=32][64]
// ---------------------------------------------------------------------------
__global__ void conv_kernel(const float* __restrict__ in, const float* __restrict__ w,
                            const float* __restrict__ bias, bf16_t* __restrict__ X0) {
  int idx = blockIdx.x * blockDim.x + threadIdx.x;
  if (idx >= TLEN * BATCH * 64) return;
  int c = idx & 63;
  int b = (idx >> 6) & 31;
  int t = idx >> 11;
  const float* ip = in + (size_t)b * 40 * 2052 + 2 * t;
  const float* wp = w + (size_t)c * 240;
  float acc = bias[c];
  for (int h = 0; h < 40; ++h)
    for (int k = 0; k < 6; ++k)
      acc += ip[(size_t)h * 2052 + k] * wp[h * 6 + k];
  X0[idx] = (bf16_t)acc;
}

// ---------------------------------------------------------------------------
// input projections: xh = X@Wh + bh ; xt = X@Wt + bt  (fp32 out, [32768,1024])
// grid (2048, 16), block 128 (4 waves), wave = one 16x16 tile for both mats
// ---------------------------------------------------------------------------
__global__ void __launch_bounds__(128) inproj_kernel(
    const bf16_t* __restrict__ X, int ldk,
    const bf16_t* __restrict__ Wh_p, const bf16_t* __restrict__ Wt_p,
    const float* __restrict__ bh, const float* __restrict__ bt,
    float* __restrict__ xh, float* __restrict__ xt) {
  const int lane   = threadIdx.x & 31;
  const int m_tile = blockIdx.x;
  const int n_tile = blockIdx.y * 4 + (threadIdx.x >> 5);
  const int nkb    = ldk / 32;
  v8f ch = zero8(), ct = zero8();
  // 1-deep software pipeline: WMMA consumes the previous iteration's loads
  v16bf a_c  = load_A_g(X, ldk, m_tile, 0);
  v16bf wh_c = load_B_p(Wh_p, n_tile, 0, nkb);
  v16bf wt_c = load_B_p(Wt_p, n_tile, 0, nkb);
  for (int kb = 0; kb < nkb; ++kb) {
    const int kn = (kb + 1 < nkb) ? kb + 1 : kb;
    v16bf a_n  = load_A_g(X, ldk, m_tile, kn);
    v16bf wh_n = load_B_p(Wh_p, n_tile, kn, nkb);
    v16bf wt_n = load_B_p(Wt_p, n_tile, kn, nkb);
    ch = wmma_bf16(a_c, wh_c, ch);
    ct = wmma_bf16(a_c, wt_c, ct);
    a_c = a_n; wh_c = wh_n; wt_c = wt_n;
  }
  const int col    = n_tile * 16 + (lane & 15);
  const int rowOff = m_tile * 16 + ((lane & 16) ? 8 : 0);
  const float bhv = bh[col], btv = bt[col];
  for (int r = 0; r < 8; ++r) {
    size_t o = (size_t)(rowOff + r) * HID + col;
    xh[o] = ch[r] + bhv;
    xt[o] = ct[r] + btv;
  }
}

// ---------------------------------------------------------------------------
// persistent recurrent scan: one layer, T=1024 steps x DEPTH=2 sub-steps
// ---------------------------------------------------------------------------
__global__ void __launch_bounds__(128, 1) rhn_scan_kernel(
    const bf16_t* __restrict__ Rh_p, const bf16_t* __restrict__ Rt_p,
    const float* __restrict__ rbh, const float* __restrict__ rbt,
    const float* __restrict__ xh,  const float* __restrict__ xt,
    bf16_t* __restrict__ ys, bf16_t* __restrict__ s_bf,
    unsigned* __restrict__ bar) {
  __shared__ bf16_t sA[BATCH * HID];          // 64 KB bf16 state stage
  const int tid    = threadIdx.x;
  const int lane   = tid & 31;
  const int wgid   = blockIdx.x * 4 + (tid >> 5);   // 0..127 global wave id
  const int m_tile = wgid & 1;                      // 2 M tiles (B=32)
  const int n_tile = wgid >> 1;                     // 64 N tiles (HID=1024)
  const int col    = n_tile * 16 + (lane & 15);
  const int rowOff = m_tile * 16 + ((lane & 16) ? 8 : 0);
  const int NKB    = HID / 32;

  v8f s_reg = zero8();                              // fp32 state slice, in regs
  for (int r = 0; r < 8; ++r) s_bf[(size_t)(rowOff + r) * HID + col] = (bf16_t)0.f;
  // real zero-init of the LDS stage (s0 == 0); also guarantees the compiler
  // sees sA as written
  {
    uint4 z; z.x = z.y = z.z = z.w = 0u;
    for (int i = tid; i < (BATCH * HID) / 8; i += 128) ((uint4*)sA)[i] = z;
  }

  float bH[DEPTH], bT[DEPTH];
  for (int l = 0; l < DEPTH; ++l) {
    bH[l] = rbh[l * HID + col];
    bT[l] = rbt[l * HID + col];
  }
  __threadfence();
  grid_sync(bar);

  const size_t mstride = (size_t)HID * HID;         // halfs per depth matrix
  for (int t = 0; t < TLEN; ++t) {
    for (int l = 0; l < DEPTH; ++l) {
      const bf16_t* Bh = Rh_p + (size_t)l * mstride;
      const bf16_t* Bt = Rt_p + (size_t)l * mstride;

#if USE_TDM
      // async state broadcast into LDS via the Tensor Data Mover; overlap the
      // DMA with the first weight-fragment loads (which don't touch LDS)
      if ((tid >> 5) == 0) tdm_stage_state(s_bf);
      v16bf wh0 = load_B_p(Bh, n_tile, 0, NKB);
      v16bf wt0 = load_B_p(Bt, n_tile, 0, NKB);
      if ((tid >> 5) == 0) __builtin_amdgcn_s_wait_tensorcnt((short)0);
      __syncthreads();
#else
      for (int i = tid; i < (BATCH * HID) / 8; i += 128)
        ((uint4*)sA)[i] = ((const uint4*)s_bf)[i];
      __syncthreads();
      v16bf wh0 = load_B_p(Bh, n_tile, 0, NKB);
      v16bf wt0 = load_B_p(Bt, n_tile, 0, NKB);
#endif
      // acquire the DMA-written LDS (escape + memory clobber): the A-fragment
      // ds-loads below must not be folded away
      const bf16_t* sAr = lds_acquire(sA);

      // 2-deep software pipeline on the sequential critical path
      v16bf aP[2], hP[2], tP[2];
      aP[0] = load_A_lds(sAr, m_tile, 0);
      hP[0] = wh0; tP[0] = wt0;
      aP[1] = load_A_lds(sAr, m_tile, 1);
      hP[1] = load_B_p(Bh, n_tile, 1, NKB);
      tP[1] = load_B_p(Bt, n_tile, 1, NKB);
      v8f ch = zero8(), ct = zero8();
#pragma unroll 4
      for (int kb = 0; kb < NKB; ++kb) {
        const int kn = (kb + 2 < NKB) ? kb + 2 : NKB - 1;
        const int slot = kb & 1;
        int pk = kb + 8;
        if (pk < NKB) {
          __builtin_prefetch(Bh + (((size_t)n_tile * NKB + pk) * 32 + lane) * 16, 0, 0);
          __builtin_prefetch(Bt + (((size_t)n_tile * NKB + pk) * 32 + lane) * 16, 0, 0);
        }
        v16bf a_u = aP[slot], h_u = hP[slot], t_u = tP[slot];
        aP[slot] = load_A_lds(sAr, m_tile, kn);
        hP[slot] = load_B_p(Bh, n_tile, kn, NKB);
        tP[slot] = load_B_p(Bt, n_tile, kn, NKB);
        ch = wmma_bf16(a_u, h_u, ch);
        ct = wmma_bf16(a_u, t_u, ct);
      }
      // highway update (each element owned by exactly one wave)
      for (int r = 0; r < 8; ++r) {
        float ph = ch[r] + bH[l];
        float pt = ct[r] + bT[l];
        if (l == 0) {
          size_t o = ((size_t)t * BATCH + rowOff + r) * HID + col;
          ph += xh[o];
          pt += xt[o];
        }
        float hv = fast_tanh(ph);
        float tv = fast_sigmoid(pt);
        float sv = s_reg[r];
        sv = hv * tv + sv * (1.f - tv);
        s_reg[r] = sv;
        s_bf[(size_t)(rowOff + r) * HID + col] = (bf16_t)sv;
      }
      __threadfence();
      grid_sync(bar);
    }
    for (int r = 0; r < 8; ++r)
      ys[((size_t)t * BATCH + rowOff + r) * HID + col] = (bf16_t)s_reg[r];
  }
}

// ---------------------------------------------------------------------------
// attention GEMM1: U = tanh(X @ W1 + b1), bf16 [32768,128]. grid (2048,2)
// ---------------------------------------------------------------------------
__global__ void __launch_bounds__(128) att1_kernel(
    const bf16_t* __restrict__ X, const bf16_t* __restrict__ W1_p,
    const float* __restrict__ b1, bf16_t* __restrict__ U) {
  const int lane   = threadIdx.x & 31;
  const int m_tile = blockIdx.x;
  const int n_tile = blockIdx.y * 4 + (threadIdx.x >> 5);   // 0..7
  const int nkb = HID / 32;
  v8f c = zero8();
  v16bf a_c = load_A_g(X, HID, m_tile, 0);
  v16bf w_c = load_B_p(W1_p, n_tile, 0, nkb);
  for (int kb = 0; kb < nkb; ++kb) {
    const int kn = (kb + 1 < nkb) ? kb + 1 : kb;
    v16bf a_n = load_A_g(X, HID, m_tile, kn);
    v16bf w_n = load_B_p(W1_p, n_tile, kn, nkb);
    c = wmma_bf16(a_c, w_c, c);
    a_c = a_n; w_c = w_n;
  }
  const int col    = n_tile * 16 + (lane & 15);
  const int rowOff = m_tile * 16 + ((lane & 16) ? 8 : 0);
  const float bv = b1[col];
  for (int r = 0; r < 8; ++r)
    U[(size_t)(rowOff + r) * 128 + col] = (bf16_t)fast_tanh(c[r] + bv);
}

// ---------------------------------------------------------------------------
// attention GEMM2: E = U @ W2 + b2, fp32 [32768,1024]. grid (2048,16)
// ---------------------------------------------------------------------------
__global__ void __launch_bounds__(128) att2_kernel(
    const bf16_t* __restrict__ U, const bf16_t* __restrict__ W2_p,
    const float* __restrict__ b2, float* __restrict__ E) {
  const int lane   = threadIdx.x & 31;
  const int m_tile = blockIdx.x;
  const int n_tile = blockIdx.y * 4 + (threadIdx.x >> 5);   // 0..63
  const int nkb = 128 / 32;
  v8f c = zero8();
  v16bf a_c = load_A_g(U, 128, m_tile, 0);
  v16bf w_c = load_B_p(W2_p, n_tile, 0, nkb);
  for (int kb = 0; kb < nkb; ++kb) {
    const int kn = (kb + 1 < nkb) ? kb + 1 : kb;
    v16bf a_n = load_A_g(U, 128, m_tile, kn);
    v16bf w_n = load_B_p(W2_p, n_tile, kn, nkb);
    c = wmma_bf16(a_c, w_c, c);
    a_c = a_n; w_c = w_n;
  }
  const int col    = n_tile * 16 + (lane & 15);
  const int rowOff = m_tile * 16 + ((lane & 16) ? 8 : 0);
  const float bv = b2[col];
  for (int r = 0; r < 8; ++r)
    E[(size_t)(rowOff + r) * HID + col] = c[r] + bv;
}

// ---------------------------------------------------------------------------
// softmax over T (online) + weighted sum: out[b,n] = sum_t alpha * x
// ---------------------------------------------------------------------------
__global__ void att_softmax_kernel(const float* __restrict__ E,
                                   const bf16_t* __restrict__ X,
                                   float* __restrict__ out) {
  int j = blockIdx.x * blockDim.x + threadIdx.x;
  if (j >= BATCH * HID) return;
  int b = j >> 10, n = j & (HID - 1);
  float m = -3.4e38f, d = 0.f, acc = 0.f;
  for (int t = 0; t < TLEN; ++t) {
    size_t o = ((size_t)t * BATCH + b) * HID + n;
    float ev = E[o];
    float xv = (float)X[o];
    if (ev > m) {
      float cv = __expf(m - ev);
      d *= cv; acc *= cv; m = ev;
    }
    float w = __expf(ev - m);
    d += w; acc += w * xv;
  }
  out[j] = acc / d;
}

// ---------------------------------------------------------------------------
static inline void pack_launch(const float* src, bf16_t* dst, int K, int N,
                               hipStream_t s) {
  int total = (K / 32) * (N / 16) * 32;
  pack_b_kernel<<<(total + 255) / 256, 256, 0, s>>>(src, dst, K, N);
}

extern "C" void kernel_launch(void* const* d_in, const int* in_sizes, int n_in,
                              void* d_out, int out_size, void* d_ws, size_t ws_size,
                              hipStream_t stream) {
  (void)in_sizes; (void)n_in; (void)out_size; (void)ws_size;
  char* ws = (char*)d_ws;
  auto alloc = [&](size_t bytes) {
    char* p = ws; ws += (bytes + 255) & ~(size_t)255; return p;
  };

  unsigned* bar = (unsigned*)alloc(1024);
  bf16_t* X0  = (bf16_t*)alloc((size_t)TLEN * BATCH * 64 * 2);
  float*  XH  = (float*) alloc((size_t)TLEN * BATCH * HID * 4);
  float*  XT  = (float*) alloc((size_t)TLEN * BATCH * HID * 4);
  bf16_t* YS0 = (bf16_t*)alloc((size_t)TLEN * BATCH * HID * 2);
  bf16_t* YS1 = (bf16_t*)alloc((size_t)TLEN * BATCH * HID * 2);
  bf16_t* SBF = (bf16_t*)alloc((size_t)BATCH * HID * 2);
  bf16_t* pW1 = (bf16_t*)alloc((size_t)HID * 128 * 2);
  bf16_t* pW2 = (bf16_t*)alloc((size_t)128 * HID * 2);
  static const int insz[4] = {64, HID, HID, HID};
  bf16_t *pWh[4], *pWt[4], *pRh[4], *pRt[4];
  for (int L = 0; L < 4; ++L) {
    pWh[L] = (bf16_t*)alloc((size_t)insz[L] * HID * 2);
    pWt[L] = (bf16_t*)alloc((size_t)insz[L] * HID * 2);
    pRh[L] = (bf16_t*)alloc((size_t)DEPTH * HID * HID * 2);
    pRt[L] = (bf16_t*)alloc((size_t)DEPTH * HID * HID * 2);
  }
  (void)alloc(4096);  // tail pad for pipelined over-reads
  // dead-buffer reuse after the last scan: E over XH, U over XT
  float*  E = XH;
  bf16_t* U = (bf16_t*)XT;

  hipMemsetAsync(bar, 0, 1024, stream);

  // pack all weights fp32 -> bf16 fragment tiles (stays resident in 192MB L2)
  for (int L = 0; L < 4; ++L) {
    const float* Wh = (const float*)d_in[3 + L * 8 + 0];
    const float* Wt = (const float*)d_in[3 + L * 8 + 1];
    const float* Rh = (const float*)d_in[3 + L * 8 + 4];
    const float* Rt = (const float*)d_in[3 + L * 8 + 5];
    pack_launch(Wh, pWh[L], insz[L], HID, stream);
    pack_launch(Wt, pWt[L], insz[L], HID, stream);
    for (int l = 0; l < DEPTH; ++l) {
      pack_launch(Rh + (size_t)l * HID * HID, pRh[L] + (size_t)l * HID * HID, HID, HID, stream);
      pack_launch(Rt + (size_t)l * HID * HID, pRt[L] + (size_t)l * HID * HID, HID, HID, stream);
    }
  }
  pack_launch((const float*)d_in[35], pW1, HID, 128, stream);
  pack_launch((const float*)d_in[37], pW2, 128, HID, stream);

  {
    int total = TLEN * BATCH * 64;
    conv_kernel<<<(total + 255) / 256, 256, 0, stream>>>(
        (const float*)d_in[0], (const float*)d_in[1], (const float*)d_in[2], X0);
  }

  const bf16_t* xin = X0;
  bf16_t* ysbuf[4] = {YS0, YS1, YS0, YS1};
  for (int L = 0; L < 4; ++L) {
    const float* bh  = (const float*)d_in[3 + L * 8 + 2];
    const float* bt  = (const float*)d_in[3 + L * 8 + 3];
    const float* rbh = (const float*)d_in[3 + L * 8 + 6];
    const float* rbt = (const float*)d_in[3 + L * 8 + 7];
    inproj_kernel<<<dim3(2048, 16), 128, 0, stream>>>(
        xin, insz[L], pWh[L], pWt[L], bh, bt, XH, XT);
    rhn_scan_kernel<<<SCAN_BLOCKS, 128, 0, stream>>>(
        pRh[L], pRt[L], rbh, rbt, XH, XT, ysbuf[L], SBF, bar + L * 16);
    xin = ysbuf[L];
  }

  att1_kernel<<<dim3(2048, 2), 128, 0, stream>>>(
      xin, pW1, (const float*)d_in[36], U);
  att2_kernel<<<dim3(2048, 16), 128, 0, stream>>>(
      U, pW2, (const float*)d_in[38], E);
  att_softmax_kernel<<<(BATCH * HID + 255) / 256, 256, 0, stream>>>(
      E, xin, (float*)d_out);
}
